// Attention_40286793237093
// MI455X (gfx1250) — matrix operand order repaired and checked by reading, not compiled, not run on hardware
//
#include <hip/hip_runtime.h>
#include <hip/hip_bf16.h>

// ---------------- problem constants (match reference) ----------------
#define BB        2
#define S_Q       1024
#define HID       4096
#define HH        32
#define DD        128
#define BLOCK     64
#define NBLK      32
#define NUM_BLOCKS (BB * NBLK)
#define S_KV      (NBLK * BLOCK)     // 2048
#define TT        (BB * S_Q)         // 2048
#define CACHE_ELEMS ((size_t)NUM_BLOCKS * BLOCK * HH * DD) // 16,777,216

typedef unsigned short u16;
typedef __attribute__((ext_vector_type(16))) __bf16 v16bf;
typedef __attribute__((ext_vector_type(8)))  float  v8f;

union FragU { v16bf v; uint4 u[2]; };

__device__ __forceinline__ u16 f2bf(float f) {
  unsigned int u = __float_as_uint(f);
  unsigned int r = u + 0x7FFFu + ((u >> 16) & 1u); // RNE
  return (u16)(r >> 16);
}

// ---- CDNA5 async global->LDS copy (ASYNCcnt), GVS addressing --------
// mem = SGPR_base + VGPR_voffset ; LDS[vdst_lds_addr] = 16 bytes
__device__ __forceinline__ void async_b128(unsigned lds_addr, unsigned voff, const void* sbase) {
  asm volatile("global_load_async_to_lds_b128 %0, %1, %2"
               :: "v"(lds_addr), "v"(voff), "s"(sbase) : "memory");
}
__device__ __forceinline__ void wait_async0() {
  asm volatile("s_wait_asynccnt 0" ::: "memory");
}
__device__ __forceinline__ unsigned lds_off(const void* p) {
  return (unsigned)(unsigned long long)p;   // low 32 bits of generic LDS addr = LDS offset
}

// ---- WMMA fragment loads (ISA 7.12.2, wave32) -----------------------
// A 16x32 bf16: lane L row = L%16. lanes<16 hold K{0..7,16..23}; lanes>=16 K{8..15,24..31}
__device__ __forceinline__ v16bf load_frag_a(const u16* row, int lane) {
  const int base = (lane < 16) ? 0 : 8;           // elements
  FragU f;
  f.u[0] = *(const uint4*)(row + base);           // K 0..7   / 8..15
  f.u[1] = *(const uint4*)(row + base + 16);      // K 16..23 / 24..31
  return f.v;
}
// B 32x16 bf16 stored as [N][K] rows: lane L col N = L%16. lanes<16 K{0..7,16..23}, lanes>=16 K{8..15,24..31}
__device__ __forceinline__ v16bf load_frag_b(const u16* row, int lane) {
  const int base = (lane < 16) ? 0 : 8;           // elements
  FragU f;
  f.u[0] = *(const uint4*)(row + base);
  f.u[1] = *(const uint4*)(row + base + 16);
  return f.v;
}

// ---------------- fp32 -> bf16 converts ------------------------------
__global__ void cvt_rm_kernel(const float* __restrict__ in, u16* __restrict__ out, size_t n) {
  size_t i = (size_t)blockIdx.x * blockDim.x + threadIdx.x;
  size_t stride = (size_t)gridDim.x * blockDim.x;
  for (; i < n; i += stride) out[i] = f2bf(in[i]);
}
// in: [R][C] fp32 -> out: [C][R] bf16 (transpose)
__global__ void cvt_tr_kernel(const float* __restrict__ in, u16* __restrict__ out, int R, int C) {
  size_t n = (size_t)R * C;
  size_t i = (size_t)blockIdx.x * blockDim.x + threadIdx.x;
  size_t stride = (size_t)gridDim.x * blockDim.x;
  for (; i < n; i += stride) {
    size_t c = i / R, r = i % R;
    out[i] = f2bf(in[r * (size_t)C + c]);
  }
}

// ---------------- bf16 WMMA GEMM: C[M][N] = A[M][K] * Bt[N][K]^T -----
// Workgroup tile 256x128, 8 waves in 4(m)x2(n) grid, 64x64 per wave.
// Async double-buffered LDS tiles, 1 barrier + 1 s_wait_asynccnt per K-step.
#define GP 56                         // LDS row pitch (112B: 16B aligned, conflict-free)
#define GEMM_LDS_A (256 * GP)         // elements per A buffer
#define GEMM_LDS_B (128 * GP)
#define GEMM_LDS_BYTES (2 * (GEMM_LDS_A + GEMM_LDS_B) * 2)

__global__ __launch_bounds__(256) void gemm_bf16_kernel(
    const u16* __restrict__ A, const u16* __restrict__ Bt, float* __restrict__ C,
    int M, int N, int K) {
  const int tid = threadIdx.x, wid = tid >> 5, lane = tid & 31;
  const int lhalf = lane & 15, hi = lane >> 4;
  const int m0 = blockIdx.y * 256, n0 = blockIdx.x * 128;
  const int wm = wid & 3;        // 4 strips of 64 rows
  const int wn = wid >> 2;       // 2 strips of 64 cols

  extern __shared__ char smem[];
  u16* As = (u16*)smem;                              // [2][256*GP]
  u16* Bs = (u16*)(smem + 2 * GEMM_LDS_A * 2);       // [2][128*GP]
  const unsigned ABUF = GEMM_LDS_A * 2;              // bytes per buffer
  const unsigned BBUF = GEMM_LDS_B * 2;

  // per-thread async chunk assignments (8-bf16 = 16B chunks)
  unsigned ldsA[4], vA[4], ldsB[2], vB[2];
#pragma unroll
  for (int j = 0; j < 4; ++j) {             // A: 256 rows x 4 chunks = 1024
    int c = tid + j * 256;
    int r = c >> 2, off = (c & 3) * 8;
    ldsA[j] = lds_off(As) + (unsigned)(r * GP + off) * 2u;
    vA[j]   = (unsigned)(r * K + off) * 2u;
  }
#pragma unroll
  for (int j = 0; j < 2; ++j) {             // B: 128 rows x 4 chunks = 512
    int c = tid + j * 256;
    int r = c >> 2, off = (c & 3) * 8;
    ldsB[j] = lds_off(Bs) + (unsigned)(r * GP + off) * 2u;
    vB[j]   = (unsigned)(r * K + off) * 2u;
  }
  const char* sA = (const char*)(A  + (size_t)m0 * K);  // advances 64B per K-step
  const char* sB = (const char*)(Bt + (size_t)n0 * K);

  v8f zero = {};
  v8f acc[4][4];
#pragma unroll
  for (int sm = 0; sm < 4; ++sm)
#pragma unroll
    for (int sn = 0; sn < 4; ++sn) acc[sm][sn] = zero;

  // prologue: tile 0 -> buffer 0
#pragma unroll
  for (int j = 0; j < 4; ++j) async_b128(ldsA[j], vA[j], sA);
#pragma unroll
  for (int j = 0; j < 2; ++j) async_b128(ldsB[j], vB[j], sB);

  const int niter = K >> 5;
  for (int it = 0; it < niter; ++it) {
    const int cur = it & 1;
    wait_async0();        // own loads for tile 'it' complete
    __syncthreads();      // everyone's loads done; everyone done reading other buffer
    if (it + 1 < niter) {
      sA += 64; sB += 64;
      const unsigned ao = (unsigned)(1 - cur) * ABUF, bo = (unsigned)(1 - cur) * BBUF;
#pragma unroll
      for (int j = 0; j < 4; ++j) async_b128(ldsA[j] + ao, vA[j], sA);
#pragma unroll
      for (int j = 0; j < 2; ++j) async_b128(ldsB[j] + bo, vB[j], sB);
    }
    const u16* Ab = As + cur * GEMM_LDS_A;
    const u16* Bb = Bs + cur * GEMM_LDS_B;
    v16bf a[4];
#pragma unroll
    for (int sm = 0; sm < 4; ++sm)
      a[sm] = load_frag_a(&Ab[(wm * 64 + sm * 16 + lhalf) * GP], lane);
#pragma unroll
    for (int sn = 0; sn < 4; ++sn) {
      v16bf bf = load_frag_b(&Bb[(wn * 64 + sn * 16 + lhalf) * GP], lane);
#pragma unroll
      for (int sm = 0; sm < 4; ++sm)
        acc[sm][sn] = __builtin_amdgcn_wmma_f32_16x16x32_bf16(
            false, a[sm], false, bf, (short)0, acc[sm][sn], false, false);
    }
  }
  // C layout: VGPR v -> row v (lanes<16) / v+8 (lanes>=16), col = lane%16
#pragma unroll
  for (int sm = 0; sm < 4; ++sm)
#pragma unroll
    for (int sn = 0; sn < 4; ++sn)
#pragma unroll
      for (int v = 0; v < 8; ++v) {
        int row = m0 + wm * 64 + sm * 16 + v + 8 * hi;
        int col = n0 + wn * 64 + sn * 16 + lhalf;
        C[(size_t)row * N + col] = acc[sm][sn][v];
      }
}

// ---------------- RoPE + scatter new k/v into output caches ----------
__global__ __launch_bounds__(64) void rope_scatter_kernel(
    const float* __restrict__ qkv, const int* __restrict__ block_offsets,
    const int* __restrict__ position_ids, const int* __restrict__ q_seqlens,
    const int* __restrict__ kv_seqlens, u16* __restrict__ q_bf,
    float* __restrict__ key_cache, float* __restrict__ value_cache) {
  const int t = blockIdx.x;        // token
  const int d = threadIdx.x;       // 0..63 (pairs with d+64)
  const int b = t / S_Q, i = t % S_Q;
  const int hist = kv_seqlens[b] - q_seqlens[b];
  const int pos = hist + i;
  const int blk = block_offsets[b * NBLK + pos / BLOCK];
  const int slot = pos % BLOCK;
  const float pid = (float)position_ids[t];
  const float inv = powf(10000.0f, -((float)d) / 64.0f);
  const float fr = pid * inv;
  const float c = cosf(fr), s = sinf(fr);
  const float* row = qkv + (size_t)t * (3 * HID);
  for (int h = 0; h < HH; ++h) {
    float q1 = row[h * DD + d], q2 = row[h * DD + d + 64];
    size_t qi = (((size_t)(b * HH + h)) * S_Q + i) * DD + d;
    q_bf[qi]      = f2bf(q1 * c - q2 * s);
    q_bf[qi + 64] = f2bf(q2 * c + q1 * s);
    float k1 = row[HID + h * DD + d], k2 = row[HID + h * DD + d + 64];
    size_t ki = (((size_t)(blk * BLOCK + slot)) * HH + h) * DD + d;
    key_cache[ki]      = k1 * c - k2 * s;
    key_cache[ki + 64] = k2 * c + k1 * s;
    value_cache[ki]      = row[2 * HID + h * DD + d];
    value_cache[ki + 64] = row[2 * HID + h * DD + d + 64];
  }
}

// ---------------- gather paged caches -> contiguous bf16 -------------
// k_bf: [b][h][s][d] ; vt_bf: [b][h][d][s] (pre-transposed for PV B-matrix)
__global__ __launch_bounds__(128) void gather_kv_kernel(
    const float* __restrict__ key_cache, const float* __restrict__ value_cache,
    const int* __restrict__ block_offsets, u16* __restrict__ k_bf, u16* __restrict__ vt_bf) {
  const int g = blockIdx.x;             // b*S_KV + s
  const int b = g / S_KV, s = g % S_KV;
  const int blk = block_offsets[b * NBLK + s / BLOCK];
  const int slot = s % BLOCK;
  const int d = threadIdx.x;            // 0..127
  const float* kc = key_cache   + ((size_t)(blk * BLOCK + slot)) * HH * DD;
  const float* vc = value_cache + ((size_t)(blk * BLOCK + slot)) * HH * DD;
  for (int h = 0; h < HH; ++h) {
    k_bf [(((size_t)(b * HH + h)) * S_KV + s) * DD + d] = f2bf(kc[h * DD + d]);
    vt_bf[(((size_t)(b * HH + h)) * DD + d) * S_KV + s] = f2bf(vc[h * DD + d]);
  }
}

// ---------------- flash attention (bf16 WMMA, fp32 softmax) ----------
// 4 waves/WG, each wave owns 16 q rows. K/V tiles async double-buffered.
#define QP 136   // Q/K LDS pitch (row 128 elems + pad, 16B aligned, conflict-free)
#define VP 72    // V^T / P LDS pitch (row 64 elems + pad)
#define ATT_Q   (64 * QP)                   // elements
#define ATT_K   (64 * QP)
#define ATT_V   (128 * VP)
#define ATT_P   (4 * 16 * VP)
#define ATT_LDS_BYTES ((ATT_Q + 2 * ATT_K + 2 * ATT_V + ATT_P) * 2)

__global__ __launch_bounds__(128) void attn_kernel(
    const u16* __restrict__ q_bf, const u16* __restrict__ k_bf, const u16* __restrict__ vt_bf,
    const int* __restrict__ q_seqlens, const int* __restrict__ kv_seqlens,
    u16* __restrict__ attn_bf) {
  const int qt = blockIdx.x;   // q tile of 64 (16 tiles)
  const int h  = blockIdx.y;
  const int b  = blockIdx.z;
  const int tid = threadIdx.x, wid = tid >> 5, lane = tid & 31;
  const int lhalf = lane & 15, hi = lane >> 4;

  extern __shared__ char smem[];
  u16* Qs = (u16*)smem;                                   // [64][QP]
  u16* Ks = (u16*)(smem + ATT_Q * 2);                     // [2][64][QP]
  u16* Vs = (u16*)(smem + (ATT_Q + 2 * ATT_K) * 2);       // [2][128][VP]
  u16* Ps = (u16*)(smem + (ATT_Q + 2 * ATT_K + 2 * ATT_V) * 2); // [4][16][VP]
  const unsigned KBUF = ATT_K * 2;   // bytes
  const unsigned VBUF = ATT_V * 2;

  const int hist = kv_seqlens[b] - q_seqlens[b];
  const int kv_len = kv_seqlens[b];

  // per-thread async chunk assignments
  unsigned ldsQ[8], vQ[8], ldsK[8], vK[8], ldsV[8], vV[8];
#pragma unroll
  for (int j = 0; j < 8; ++j) {            // Q/K: 64 rows x 16 chunks
    int idx = tid + j * 128;
    int r = idx >> 4, c = (idx & 15) * 8;
    ldsQ[j] = lds_off(Qs) + (unsigned)(r * QP + c) * 2u;
    ldsK[j] = lds_off(Ks) + (unsigned)(r * QP + c) * 2u;
    vQ[j] = vK[j] = (unsigned)(r * DD + c) * 2u;
  }
#pragma unroll
  for (int j = 0; j < 8; ++j) {            // V^T: 128 rows x 8 chunks
    int idx = tid + j * 128;
    int r = idx >> 3, c = (idx & 7) * 8;
    ldsV[j] = lds_off(Vs) + (unsigned)(r * VP + c) * 2u;
    vV[j]   = (unsigned)(r * S_KV + c) * 2u;
  }
  const char* sQ = (const char*)(q_bf + (((size_t)(b * HH + h)) * S_Q + qt * 64) * DD);
  const char* sK = (const char*)(k_bf + ((size_t)(b * HH + h)) * S_KV * DD); // +16384B per kv tile
  const char* sV = (const char*)(vt_bf + ((size_t)(b * HH + h)) * DD * S_KV); // +128B per kv tile

  v8f zero = {};
  v8f o[8];
#pragma unroll
  for (int n = 0; n < 8; ++n) o[n] = zero;
  float mrow[8], lrow[8];
#pragma unroll
  for (int v = 0; v < 8; ++v) { mrow[v] = -1e30f; lrow[v] = 0.0f; }

  const int last_kpos = hist + qt * 64 + 63;
  int tmax = (last_kpos >> 6) + 1;
  int tcap = (kv_len + 63) >> 6;
  if (tcap < tmax) tmax = tcap;
  const float scale = 0.08838834764831845f; // 1/sqrt(128)

  // prologue: Q + KV tile 0 -> buffer 0
#pragma unroll
  for (int j = 0; j < 8; ++j) async_b128(ldsQ[j], vQ[j], sQ);
#pragma unroll
  for (int j = 0; j < 8; ++j) async_b128(ldsK[j], vK[j], sK);
#pragma unroll
  for (int j = 0; j < 8; ++j) async_b128(ldsV[j], vV[j], sV);

  for (int t = 0; t < tmax; ++t) {
    const int cur = t & 1;
    wait_async0();
    __syncthreads();
    if (t + 1 < tmax) {
      sK += 64 * DD * 2;   // next 64 kv rows
      sV += 64 * 2;        // next 64 kv cols
      const unsigned ko = (unsigned)(1 - cur) * KBUF, vo = (unsigned)(1 - cur) * VBUF;
#pragma unroll
      for (int j = 0; j < 8; ++j) async_b128(ldsK[j] + ko, vK[j], sK);
#pragma unroll
      for (int j = 0; j < 8; ++j) async_b128(ldsV[j] + vo, vV[j], sV);
    }
    const u16* Kb = Ks + cur * ATT_K;
    const u16* Vb = Vs + cur * ATT_V;

    // ---- S = Q * K^T (16 q-rows per wave x 64 kv) ----
    v8f s[4];
#pragma unroll
    for (int nt = 0; nt < 4; ++nt) s[nt] = zero;
#pragma unroll
    for (int ks = 0; ks < 4; ++ks) {
      v16bf a = load_frag_a(&Qs[(wid * 16 + lhalf) * QP + ks * 32], lane);
#pragma unroll
      for (int nt = 0; nt < 4; ++nt) {
        v16bf bf = load_frag_b(&Kb[(nt * 16 + lhalf) * QP + ks * 32], lane);
        s[nt] = __builtin_amdgcn_wmma_f32_16x16x32_bf16(
            false, a, false, bf, (short)0, s[nt], false, false);
      }
    }
    // ---- scale + causal mask ----
    const bool need_mask = (t * 64 + 63 > hist + qt * 64) || (t * 64 + 63 >= kv_len);
#pragma unroll
    for (int nt = 0; nt < 4; ++nt)
#pragma unroll
      for (int v = 0; v < 8; ++v) {
        float x = s[nt][v] * scale;
        if (need_mask) {
          int kpos = t * 64 + nt * 16 + lhalf;
          int qpos = hist + qt * 64 + wid * 16 + v + 8 * hi;
          if (kpos > qpos || kpos >= kv_len) x = -1e30f;
        }
        s[nt][v] = x;
      }
    // ---- flash update ----
    float mstep[8], alpha[8], ls[8];
#pragma unroll
    for (int v = 0; v < 8; ++v) {
      float m = s[0][v];
#pragma unroll
      for (int nt = 1; nt < 4; ++nt) m = fmaxf(m, s[nt][v]);
#pragma unroll
      for (int off = 1; off < 16; off <<= 1)
        m = fmaxf(m, __shfl_xor(m, off, 32));
      mstep[v] = m;
    }
#pragma unroll
    for (int v = 0; v < 8; ++v) {
      float mn = fmaxf(mrow[v], mstep[v]);
      alpha[v] = __expf(mrow[v] - mn);
      mrow[v] = mn;
      ls[v] = 0.0f;
    }
#pragma unroll
    for (int nt = 0; nt < 4; ++nt)
#pragma unroll
      for (int v = 0; v < 8; ++v) {
        float p = __expf(s[nt][v] - mrow[v]);
        s[nt][v] = p;
        ls[v] += p;
      }
#pragma unroll
    for (int v = 0; v < 8; ++v) {
      float l = ls[v];
#pragma unroll
      for (int off = 1; off < 16; off <<= 1) l += __shfl_xor(l, off, 32);
      lrow[v] = lrow[v] * alpha[v] + l;
    }
#pragma unroll
    for (int n = 0; n < 8; ++n)
#pragma unroll
      for (int v = 0; v < 8; ++v) o[n][v] *= alpha[v];

    // ---- P (C layout) -> wave-private LDS (row-major, A layout source) ----
    u16* pb = &Ps[wid * 16 * VP];
#pragma unroll
    for (int nt = 0; nt < 4; ++nt)
#pragma unroll
      for (int v = 0; v < 8; ++v)
        pb[(v + 8 * hi) * VP + nt * 16 + lhalf] = f2bf(s[nt][v]);
    __builtin_amdgcn_wave_barrier();   // same-wave LDS ops are in-order

    // ---- O += P * V ----
#pragma unroll
    for (int ks = 0; ks < 2; ++ks) {
      v16bf a = load_frag_a(&pb[lhalf * VP + ks * 32], lane);
#pragma unroll
      for (int nt = 0; nt < 8; ++nt) {
        v16bf bf = load_frag_b(&Vb[(nt * 16 + lhalf) * VP + ks * 32], lane);
        o[nt] = __builtin_amdgcn_wmma_f32_16x16x32_bf16(
            false, a, false, bf, (short)0, o[nt], false, false);
      }
    }
  }
  // ---- epilogue: divide by l, write attn (T, H*D) as bf16 ----
#pragma unroll
  for (int nt = 0; nt < 8; ++nt)
#pragma unroll
    for (int v = 0; v < 8; ++v) {
      float val = o[nt][v] / lrow[v];
      int qrow = qt * 64 + wid * 16 + v + 8 * hi;
      int dcol = nt * 16 + lhalf;
      attn_bf[((size_t)(b * S_Q + qrow)) * HID + h * DD + dcol] = f2bf(val);
    }
}

// ---------------- host-side pipeline ---------------------------------
extern "C" void kernel_launch(void* const* d_in, const int* in_sizes, int n_in,
                              void* d_out, int out_size, void* d_ws, size_t ws_size,
                              hipStream_t stream) {
  (void)in_sizes; (void)n_in; (void)out_size;
  const float* hidden    = (const float*)d_in[0];
  const float* w_pack    = (const float*)d_in[1];
  const float* o_proj    = (const float*)d_in[2];
  const float* past_key  = (const float*)d_in[3];
  const float* past_val  = (const float*)d_in[4];
  const int* block_off   = (const int*)d_in[5];
  const int* position_ids= (const int*)d_in[6];
  const int* q_seqlens   = (const int*)d_in[7];
  const int* kv_seqlens  = (const int*)d_in[8];

  float* out        = (float*)d_out;                       // [T][HID]
  float* key_cache  = out + (size_t)TT * HID;              // [64][64][32][128]
  float* value_cache= key_cache + CACHE_ELEMS;

  // workspace carve (total ~336 MB)
  char* w = (char*)d_ws;
  u16*  A_bf   = (u16*)w;                    w += (size_t)TT * HID * 2;          // hidden bf16
  u16*  Wp_bf  = (u16*)w;                    w += (size_t)3 * HID * HID * 2;     // w_pack^T bf16 [3HID][HID]
  u16*  Op_bf  = (u16*)w;                    w += (size_t)HID * HID * 2;         // o_proj^T bf16 [HID][HID]
  float* qkv   = (float*)w;                  w += (size_t)TT * 3 * HID * 4;      // qkv fp32
  u16*  q_bf   = (u16*)w;                    w += (size_t)BB * HH * S_Q * DD * 2;
  u16*  k_bf   = (u16*)w;                    w += (size_t)BB * HH * S_KV * DD * 2;
  u16*  vt_bf  = (u16*)w;                    w += (size_t)BB * HH * DD * S_KV * 2;
  u16*  attn_bf= (u16*)w;                    w += (size_t)TT * HID * 2;
  (void)ws_size;

  // seed output caches from past, then scatter new tokens over them
  hipMemcpyAsync(key_cache,   past_key, CACHE_ELEMS * sizeof(float), hipMemcpyDeviceToDevice, stream);
  hipMemcpyAsync(value_cache, past_val, CACHE_ELEMS * sizeof(float), hipMemcpyDeviceToDevice, stream);

  // fp32 -> bf16 converts (weights transposed to [N][K])
  cvt_rm_kernel<<<2048, 256, 0, stream>>>(hidden, A_bf, (size_t)TT * HID);
  cvt_tr_kernel<<<4096, 256, 0, stream>>>(w_pack, Wp_bf, HID, 3 * HID);
  cvt_tr_kernel<<<2048, 256, 0, stream>>>(o_proj, Op_bf, HID, HID);

  // qkv = hidden @ w_pack   (2048 x 12288 x 4096)
  gemm_bf16_kernel<<<dim3(3 * HID / 128, TT / 256), 256, GEMM_LDS_BYTES, stream>>>(
      A_bf, Wp_bf, qkv, TT, 3 * HID, HID);

  // RoPE + cache scatter + q bf16
  rope_scatter_kernel<<<TT, 64, 0, stream>>>(
      qkv, block_off, position_ids, q_seqlens, kv_seqlens, q_bf, key_cache, value_cache);

  // gather paged caches into contiguous bf16 (V pre-transposed)
  gather_kv_kernel<<<BB * S_KV, 128, 0, stream>>>(key_cache, value_cache, block_off, k_bf, vt_bf);

  // flash attention
  attn_kernel<<<dim3(S_Q / 64, HH, BB), 128, ATT_LDS_BYTES, stream>>>(
      q_bf, k_bf, vt_bf, q_seqlens, kv_seqlens, attn_bf);

  // out = attn @ o_proj_w   (2048 x 4096 x 4096)
  gemm_bf16_kernel<<<dim3(HID / 128, TT / 256), 256, GEMM_LDS_BYTES, stream>>>(
      attn_bf, Op_bf, out, TT, HID, HID);
}
